// SAP_29815662969282
// MI455X (gfx1250) — matrix operand order repaired and checked by reading, not compiled, not run on hardware
//
#include <hip/hip_runtime.h>

// ---------------------------------------------------------------------------
// SAP forward for MI455X (gfx1250).
//
// All conv-shaped work runs through ONE templated implicit-GEMM kernel on
// v_wmma_f32_16x16x32_f16:
//   - 64(M) x 128(N) output tile per 256-thread block (8 wave32s, 4 accums ea)
//   - K chunks of 32; weights pre-packed f16 **t-major** ([m][t*Cin+c]) so
//     t = row>>8, c = row&255 (no integer division in the hot loop)
//   - deformable sampling: bilinear weights/indices precomputed per (b,t,px)
//     into float4/int4 tables; hot loop = 4 gathers + dot4
//   - epilogue: +bias | 2*sigmoid | *mul (fuses final y*a into last GEMM)
//
// Input ordering assumption (setup_inputs() dict insertion order, recursive):
//   0: x
//   1+7i .. 7+7i (i=0..3): off_w, off_b, mod_w, mod_b, w, bn_g, bn_b
//   29: gp_w  30: gp_bn_g  31: gp_bn_b  32: conv1_w  33: bn1_g  34: bn1_b
//   35..42: c0h_w,c0h_b,c0v_w,c0v_b,sh_w,sh_b,sv_w,sv_b   43: c1_w  44: c1_b
// ---------------------------------------------------------------------------

typedef _Float16 v16h __attribute__((ext_vector_type(16)));
typedef _Float16 v8h  __attribute__((ext_vector_type(8)));
typedef float    v8f  __attribute__((ext_vector_type(8)));

#define LDSS 40   // LDS row stride in f16 (80B: 16B aligned, bank-skewed)

// A fragment: 16x32 f16. ISA 7.12.2: lanes 0-15 -> M=lane, K {0-7,16-23};
// lanes 16-31 -> K {8-15, 24-31}.
__device__ __forceinline__ v16h load_fragA(const _Float16* tile, int lane) {
  const int half = lane >> 4;
  const int r    = lane & 15;
  const _Float16* row = tile + r * LDSS;
  v8h lo = *(const v8h*)(row + (half ? 8  : 0));
  v8h hi = *(const v8h*)(row + (half ? 24 : 16));
  return __builtin_shufflevector(lo, hi, 0,1,2,3,4,5,6,7,8,9,10,11,12,13,14,15);
}

// B fragment: 32x16 f16, stored transposed in LDS as [n][K]; lanes 0-15 hold
// K {0-7,16-23} for N=lane, lanes 16-31 hold K {8-15,24-31}.
__device__ __forceinline__ v16h load_fragB(const _Float16* tile, int lane) {
  const int half = lane >> 4;
  const int n    = lane & 15;
  const _Float16* row = tile + n * LDSS + half * 8;
  v8h lo = *(const v8h*)(row);
  v8h hi = *(const v8h*)(row + 16);
  return __builtin_shufflevector(lo, hi, 0,1,2,3,4,5,6,7,8,9,10,11,12,13,14,15);
}

// ---------------------------------------------------------------------------
// Implicit-GEMM conv.  K9 in {1,9,49}; for K9>1, Cin must be 256 (true here).
// DEFORM=false: plain conv, zero pad.  DEFORM=true: use precomputed tables.
// act: 0 = +bias ; 1 = 2*sigmoid(+bias) ; 2 = (+bias)*mul[idx]
// ---------------------------------------------------------------------------
template<int K9, int KSZ, bool DEFORM>
__global__ __launch_bounds__(256)
void k_gemm_conv(const float* __restrict__ src, const _Float16* __restrict__ wt,
                 const float* __restrict__ wtab, const int* __restrict__ itab,
                 const float* __restrict__ bias, const float* __restrict__ mul,
                 float* __restrict__ dst, int M, int Cin, int pad, int act)
{
  constexpr int H = 64, W = 64, HW = 4096;
  __shared__ _Float16 ldsA[64  * LDSS];
  __shared__ _Float16 ldsB[128 * LDSS];

  const int tid  = threadIdx.x;
  const int lane = tid & 31;
  const int wave = tid >> 5;
  const int m0    = blockIdx.x * 64;
  const int ptile = blockIdx.y * 128;        // flat pixel over B*HW (HW%128==0)
  const int b  = ptile / HW;
  const int p0 = ptile - b * HW;

  // loader lanes
  const int la_row = tid & 63;
  const int la_k   = (tid >> 6) << 3;        // 0,8,16,24
  const int lb_n   = tid & 127;
  const int lb_k   = (tid >> 7) << 4;        // 0,16
  const int px = p0 + lb_n;
  const int h  = px >> 6;
  const int w  = px & 63;

  // wave -> C subtiles: mi in 0..3 (M), 4 N subtiles starting at nh
  const int mi = wave & 3;
  const int nh = (wave >> 2) << 2;           // 0 or 4

  v8f acc[4];
  #pragma unroll
  for (int q = 0; q < 4; ++q) acc[q] = (v8f){0.f,0.f,0.f,0.f,0.f,0.f,0.f,0.f};

  const int CK  = Cin * K9;
  const int gmA = m0 + la_row;
  const bool arow_ok = gmA < M;
  const _Float16* wrow = wt + (size_t)(arow_ok ? gmA : 0) * CK;

  for (int kk = 0; kk < CK; kk += 32) {
    // ---- A tile: weights (f16, t-major packed) ----
    v8h a8 = {(_Float16)0,(_Float16)0,(_Float16)0,(_Float16)0,
              (_Float16)0,(_Float16)0,(_Float16)0,(_Float16)0};
    if (arow_ok) a8 = *(const v8h*)(wrow + kk + la_k);
    *(v8h*)(&ldsA[la_row * LDSS + la_k]) = a8;
    if (arow_ok && (kk + 32) < CK)
      __builtin_prefetch(wrow + kk + 32 + la_k, 0, 0);   // global_prefetch_b8

    // ---- B tile: sampled activations (t constant across this thread chunk) ----
    const int row0 = kk + lb_k;
    const int t = (K9 == 1) ? 0 : (row0 >> 8);           // Cin==256 when K9>1
    float4 wv; int4 iv;
    int iy = 0, ix = 0; bool valid = false;
    if (DEFORM) {
      const size_t tb = (((size_t)b * K9 + t) * HW + px) << 2;
      wv = *(const float4*)(wtab + tb);
      iv = *(const int4*)(itab + tb);
    } else {
      const int ky = t / KSZ;                // compile-time KSZ
      const int kx = t - ky * KSZ;
      iy = h - pad + ky;
      ix = w - pad + kx;
      valid = (iy >= 0) && (iy < H) && (ix >= 0) && (ix < W);
    }
    #pragma unroll
    for (int jj = 0; jj < 2; ++jj) {
      v8h b8;
      #pragma unroll
      for (int j = 0; j < 8; ++j) {
        const int row = row0 + jj * 8 + j;
        const int c = (K9 == 1) ? row : (row & 255);
        const float* xb = src + ((size_t)b * Cin + c) * HW;
        float s;
        if (DEFORM)
          s = xb[iv.x] * wv.x + xb[iv.y] * wv.y + xb[iv.z] * wv.z + xb[iv.w] * wv.w;
        else
          s = valid ? xb[iy * W + ix] : 0.f;
        b8[j] = (_Float16)s;
      }
      *(v8h*)(&ldsB[lb_n * LDSS + lb_k + jj * 8]) = b8;
    }

    __syncthreads();
    v16h af = load_fragA(&ldsA[(mi * 16) * LDSS], lane);
    #pragma unroll
    for (int q = 0; q < 4; ++q) {
      v16h bf = load_fragB(&ldsB[((nh + q) * 16) * LDSS], lane);
      acc[q] = __builtin_amdgcn_wmma_f32_16x16x32_f16(false, af, false, bf,
                                                      (short)0, acc[q], false, false);
    }
    __syncthreads();
  }

  // ---- epilogue (C/D layout: lane half selects M+8, N = lane&15) ----
  const int half = lane >> 4;
  const int nl   = lane & 15;
  #pragma unroll
  for (int r = 0; r < 8; ++r) {
    const int gm = m0 + mi * 16 + r + 8 * half;
    if (gm >= M) continue;
    const float bz = bias ? bias[gm] : 0.f;
    #pragma unroll
    for (int q = 0; q < 4; ++q) {
      float v = acc[q][r] + bz;
      const int n = (nh + q) * 16 + nl;
      const size_t idx = ((size_t)b * M + gm) * HW + p0 + n;
      if (act == 1)      v = 2.f / (1.f + __expf(-v));
      else if (act == 2) v *= mul[idx];
      dst[idx] = v;
    }
  }
}

// ---------------------------------------------------------------------------
// Weight pack: f32 OIHW [m][c][t] -> f16 t-major [m][t*Cin + c]
__global__ void k_wt_pack(const float* __restrict__ src, _Float16* __restrict__ dst,
                          int Cin, int K9, int total)
{
  int i = blockIdx.x * blockDim.x + threadIdx.x;
  if (i >= total) return;
  const int t = i % K9;
  const int c = (i / K9) % Cin;
  const int m = i / (K9 * Cin);
  dst[(size_t)m * Cin * K9 + t * Cin + c] = (_Float16)src[i];
}

// Precompute bilinear gather tables per (b,t,px): 4 clamped indices + 4
// weights (x valid x mask) — matches reference clip()*valid semantics.
__global__ void k_deform_prep(const float* __restrict__ off, const float* __restrict__ msk,
                              float* __restrict__ wtab, int* __restrict__ itab,
                              int K9, int ksz, int pad, int total /* B*K9*HW */)
{
  int i = blockIdx.x * blockDim.x + threadIdx.x;
  if (i >= total) return;
  constexpr int H = 64, W = 64, HW = 4096;
  const int px = i % HW;
  const int t  = (i / HW) % K9;
  const int b  = i / (HW * K9);
  const int h = px >> 6, w = px & 63;
  const int ky = t / ksz, kx = t - (t / ksz) * ksz;
  const float oy = off[((size_t)b * 2 * K9 + 2 * t    ) * HW + px];
  const float ox = off[((size_t)b * 2 * K9 + 2 * t + 1) * HW + px];
  const float mv = msk[i];
  const float py = (float)(h - pad + ky) + oy;
  const float qx = (float)(w - pad + kx) + ox;
  const float fy = floorf(py), fx = floorf(qx);
  const int y0 = (int)fy, x0 = (int)fx;
  const float wy = py - fy, wx = qx - fx;
  const int y1 = y0 + 1, x1 = x0 + 1;
  const float vy0 = (y0 >= 0 && y0 < H) ? 1.f : 0.f;
  const float vy1 = (y1 >= 0 && y1 < H) ? 1.f : 0.f;
  const float vx0 = (x0 >= 0 && x0 < W) ? 1.f : 0.f;
  const float vx1 = (x1 >= 0 && x1 < W) ? 1.f : 0.f;
  const int y0c = min(max(y0, 0), H - 1), y1c = min(max(y1, 0), H - 1);
  const int x0c = min(max(x0, 0), W - 1), x1c = min(max(x1, 0), W - 1);
  float4 wvv;
  wvv.x = (1.f - wy) * (1.f - wx) * vy0 * vx0 * mv;
  wvv.y = (1.f - wy) * wx         * vy0 * vx1 * mv;
  wvv.z = wy * (1.f - wx)         * vy1 * vx0 * mv;
  wvv.w = wy * wx                 * vy1 * vx1 * mv;
  int4 ivv;
  ivv.x = y0c * W + x0c;  ivv.y = y0c * W + x1c;
  ivv.z = y1c * W + x0c;  ivv.w = y1c * W + x1c;
  *(float4*)(wtab + (size_t)i * 4) = wvv;
  *(int4*)(itab + (size_t)i * 4)   = ivv;
}

// per-channel BN stats over (B,HW): scale = g*rsqrt(var+eps), shift = b - mean*scale
__global__ __launch_bounds__(256)
void k_bn_stats(const float* __restrict__ src, const float* __restrict__ gam,
                const float* __restrict__ bet, float* __restrict__ scale,
                float* __restrict__ shift, int C, int Bn, int HW)
{
  const int c = blockIdx.x;
  const int cnt = Bn * HW;
  float s = 0.f, s2 = 0.f;
  for (int i = threadIdx.x; i < cnt; i += 256) {
    const int bb = i / HW;
    const int p  = i - bb * HW;
    const float v = src[((size_t)bb * C + c) * HW + p];
    s += v; s2 += v * v;
  }
  __shared__ float sh[512];
  sh[threadIdx.x] = s; sh[256 + threadIdx.x] = s2;
  __syncthreads();
  for (int st = 128; st > 0; st >>= 1) {
    if (threadIdx.x < st) {
      sh[threadIdx.x]       += sh[threadIdx.x + st];
      sh[256 + threadIdx.x] += sh[256 + threadIdx.x + st];
    }
    __syncthreads();
  }
  if (threadIdx.x == 0) {
    const float m   = sh[0] / cnt;
    const float var = sh[256] / cnt - m * m;
    const float sc  = gam[c] * rsqrtf(var + 1e-5f);
    scale[c] = sc;
    shift[c] = bet[c] - m * sc;
  }
}

__global__ void k_bn_apply_relu(const float* __restrict__ src, const float* __restrict__ scale,
                                const float* __restrict__ shift, float* __restrict__ dst,
                                int C, int dstC, int c_off, int HW, int total)
{
  int i = blockIdx.x * blockDim.x + threadIdx.x;
  if (i >= total) return;
  const int p  = i % HW;
  const int c  = (i / HW) % C;
  const int bb = i / (HW * C);
  float v = src[i] * scale[c] + shift[c];
  v = v > 0.f ? v : 0.f;
  dst[((size_t)bb * dstC + c_off + c) * HW + p] = v;
}

__global__ __launch_bounds__(256)
void k_gp_reduce(const float* __restrict__ src, float* __restrict__ out, int HW) {
  const float* s = src + (size_t)blockIdx.x * HW;
  float acc = 0.f;
  for (int i = threadIdx.x; i < HW; i += 256) acc += s[i];
  __shared__ float sh[256];
  sh[threadIdx.x] = acc; __syncthreads();
  for (int st = 128; st > 0; st >>= 1) {
    if (threadIdx.x < st) sh[threadIdx.x] += sh[threadIdx.x + st];
    __syncthreads();
  }
  if (threadIdx.x == 0) out[blockIdx.x] = sh[0] / HW;
}

__global__ __launch_bounds__(256)
void k_gp_matvec(const float* __restrict__ wgt, const float* __restrict__ in,
                 float* __restrict__ out, int C) {
  const int b = blockIdx.x, o = threadIdx.x;
  float acc = 0.f;
  for (int c = 0; c < C; ++c) acc += wgt[o * C + c] * in[b * C + c];
  out[b * C + o] = acc;
}

// BN over batch dim + relu + broadcast into concat slot
__global__ void k_gp_bcast(const float* __restrict__ gp, const float* __restrict__ gam,
                           const float* __restrict__ bet, float* __restrict__ dst,
                           int C, int Bn, int dstC, int c_off, int HW, int total)
{
  int i = blockIdx.x * blockDim.x + threadIdx.x;
  if (i >= total) return;
  const int p  = i % HW;
  const int c  = (i / HW) % C;
  const int bb = i / (HW * C);
  float m = 0.f, s2 = 0.f;
  for (int j = 0; j < Bn; ++j) { const float v = gp[j * C + c]; m += v; s2 += v * v; }
  m /= Bn;
  const float var = s2 / Bn - m * m;
  float v = (gp[bb * C + c] - m) * rsqrtf(var + 1e-5f) * gam[c] + bet[c];
  v = v > 0.f ? v : 0.f;
  dst[((size_t)bb * dstC + c_off + c) * HW + p] = v;
}

// depthwise 3-tap conv (horiz: 1x3, vert: 3x1), zero pad = dil
__global__ void k_dw3(const float* __restrict__ src, const float* __restrict__ wgt,
                      const float* __restrict__ bias, float* __restrict__ dst,
                      int C, int H, int W, int horiz, int dil, int total)
{
  int i = blockIdx.x * blockDim.x + threadIdx.x;
  if (i >= total) return;
  const int w = i % W;
  const int h = (i / W) % H;
  const int c = (i / (W * H)) % C;
  const float* s = src + (size_t)(i / (W * H)) * (W * H);
  float acc = bias[c];
  #pragma unroll
  for (int j = 0; j < 3; ++j) {
    const int hh = h + (horiz ? 0 : (j - 1) * dil);
    const int ww = w + (horiz ? (j - 1) * dil : 0);
    if (hh >= 0 && hh < H && ww >= 0 && ww < W)
      acc += wgt[c * 3 + j] * s[hh * W + ww];
  }
  dst[i] = acc;
}

// ---------------------------------------------------------------------------
extern "C" void kernel_launch(void* const* d_in, const int* in_sizes, int n_in,
                              void* d_out, int out_size, void* d_ws, size_t ws_size,
                              hipStream_t stream)
{
  const int B = 2, C = 256, H = 64, W = 64, HW = H * W;
  const int ks[4]   = {1, 1, 3, 7};
  const int pads[4] = {0, 0, 1, 3};

  const float* x = (const float*)d_in[0];
  auto IN = [&](int i) { return (const float*)d_in[i]; };

  // bump allocator over workspace
  char* wsb = (char*)d_ws;
  size_t cur = 0;
  auto alloc = [&](size_t bytes) -> void* {
    void* p = wsb + cur;
    cur = (cur + bytes + 255) & ~(size_t)255;
    return p;
  };

  _Float16 *offw16[4], *modw16[4], *w16[4];
  for (int i = 0; i < 4; ++i) {
    const int k = ks[i], K9 = k * k, CK = C * K9;
    offw16[i] = (_Float16*)alloc((size_t)2 * K9 * CK * 2);
    modw16[i] = (_Float16*)alloc((size_t)K9 * CK * 2);
    w16[i]    = (_Float16*)alloc((size_t)C * CK * 2);
  }
  _Float16* conv1w16 = (_Float16*)alloc((size_t)C * 5 * C * 2);
  _Float16* c1w16    = (_Float16*)alloc((size_t)C * C * 2);
  float* off_buf = (float*)alloc((size_t)B * 2 * 49 * HW * 4);
  float* mod_buf = (float*)alloc((size_t)B * 49 * HW * 4);
  float* wtab    = (float*)alloc((size_t)B * 49 * HW * 4 * 4);
  int*   itab    = (int*)  alloc((size_t)B * 49 * HW * 4 * 4);
  float* preBN   = (float*)alloc((size_t)B * C * HW * 4);
  float* concat  = (float*)alloc((size_t)B * 5 * C * HW * 4);
  float* ybuf    = (float*)alloc((size_t)B * C * HW * 4);
  float* a0      = (float*)alloc((size_t)B * C * HW * 4);
  float* a1      = (float*)alloc((size_t)B * C * HW * 4);
  float* bnsc    = (float*)alloc(C * 4);
  float* bnsh    = (float*)alloc(C * 4);
  float* gp_in   = (float*)alloc(B * C * 4);
  float* gp_out  = (float*)alloc(B * C * 4);

  auto pack = [&](const float* s, _Float16* d, int M, int Cin, int K9) {
    const int n = M * Cin * K9;
    k_wt_pack<<<(n + 255) / 256, 256, 0, stream>>>(s, d, Cin, K9, n);
  };
  for (int i = 0; i < 4; ++i) {
    const int base = 1 + 7 * i;
    const int k = ks[i], K9 = k * k;
    pack(IN(base + 0), offw16[i], 2 * K9, C, K9);
    pack(IN(base + 2), modw16[i], K9, C, K9);
    pack(IN(base + 4), w16[i], C, C, K9);
  }
  pack(IN(32), conv1w16, C, 5 * C, 1);
  pack(IN(43), c1w16, C, C, 1);

  const dim3 blk(256);
  const int ntiles = B * HW / 128;   // 64
  auto gemm = [&](int k, bool deform, const _Float16* wt, const float* srcp,
                  const float* biasp, const float* mulp, float* dstp,
                  int M, int Cin, int pad, int act) {
    dim3 grid((M + 63) / 64, ntiles);
    if (deform) {
      if (k == 1)      k_gemm_conv<1, 1, true ><<<grid, blk, 0, stream>>>(srcp, wt, wtab, itab, biasp, mulp, dstp, M, Cin, pad, act);
      else if (k == 3) k_gemm_conv<9, 3, true ><<<grid, blk, 0, stream>>>(srcp, wt, wtab, itab, biasp, mulp, dstp, M, Cin, pad, act);
      else             k_gemm_conv<49,7, true ><<<grid, blk, 0, stream>>>(srcp, wt, wtab, itab, biasp, mulp, dstp, M, Cin, pad, act);
    } else {
      if (k == 1)      k_gemm_conv<1, 1, false><<<grid, blk, 0, stream>>>(srcp, wt, nullptr, nullptr, biasp, mulp, dstp, M, Cin, pad, act);
      else if (k == 3) k_gemm_conv<9, 3, false><<<grid, blk, 0, stream>>>(srcp, wt, nullptr, nullptr, biasp, mulp, dstp, M, Cin, pad, act);
      else             k_gemm_conv<49,7, false><<<grid, blk, 0, stream>>>(srcp, wt, nullptr, nullptr, biasp, mulp, dstp, M, Cin, pad, act);
    }
  };

  const int tot = B * C * HW;
  for (int i = 0; i < 4; ++i) {
    const int base = 1 + 7 * i;
    const int k = ks[i], K9 = k * k, pad = pads[i];
    // offset conv (+bias), modulation conv (+bias, 2*sigmoid)
    gemm(k, false, offw16[i], x, IN(base + 1), nullptr, off_buf, 2 * K9, C, pad, 0);
    gemm(k, false, modw16[i], x, IN(base + 3), nullptr, mod_buf, K9, C, pad, 1);
    // bilinear tables, then deformable conv GEMM
    const int ptot = B * K9 * HW;
    k_deform_prep<<<(ptot + 255) / 256, blk, 0, stream>>>(off_buf, mod_buf, wtab, itab,
                                                          K9, k, pad, ptot);
    gemm(k, true, w16[i], x, nullptr, nullptr, preBN, C, C, pad, 0);
    // BN + relu into concat slot i
    k_bn_stats<<<C, blk, 0, stream>>>(preBN, IN(base + 5), IN(base + 6), bnsc, bnsh, C, B, HW);
    k_bn_apply_relu<<<(tot + 255) / 256, blk, 0, stream>>>(preBN, bnsc, bnsh, concat,
                                                           C, 5 * C, i * C, HW, tot);
  }

  // global-pool branch -> concat slot 4
  k_gp_reduce<<<B * C, blk, 0, stream>>>(x, gp_in, HW);
  k_gp_matvec<<<B, C, 0, stream>>>(IN(29), gp_in, gp_out, C);
  k_gp_bcast<<<(tot + 255) / 256, blk, 0, stream>>>(gp_out, IN(30), IN(31), concat,
                                                    C, B, 5 * C, 4 * C, HW, tot);

  // conv1 (1x1, 1280->256) + BN + relu -> y
  gemm(1, false, conv1w16, concat, nullptr, nullptr, preBN, C, 5 * C, 0, 0);
  k_bn_stats<<<C, blk, 0, stream>>>(preBN, IN(33), IN(34), bnsc, bnsh, C, B, HW);
  k_bn_apply_relu<<<(tot + 255) / 256, blk, 0, stream>>>(preBN, bnsc, bnsh, ybuf,
                                                         C, C, 0, HW, tot);

  // attention chain (depthwise, bandwidth-bound)
  dim3 eg((tot + 255) / 256);
  k_dw3<<<eg, blk, 0, stream>>>(ybuf, IN(35), IN(36), a0, C, H, W, 1, 1, tot);
  k_dw3<<<eg, blk, 0, stream>>>(a0,   IN(37), IN(38), a1, C, H, W, 0, 1, tot);
  k_dw3<<<eg, blk, 0, stream>>>(a1,   IN(39), IN(40), a0, C, H, W, 1, 2, tot);
  k_dw3<<<eg, blk, 0, stream>>>(a0,   IN(41), IN(42), a1, C, H, W, 0, 2, tot);

  // c1 (1x1, 256->256) + bias, fused with final y*a gate -> d_out
  gemm(1, false, c1w16, a1, IN(44), ybuf, (float*)d_out, C, C, 0, 2);

  (void)in_sizes; (void)n_in; (void)out_size; (void)ws_size;
}
